// DSFD_33285996544222
// MI455X (gfx1250) — matrix operand order repaired and checked by reading, not compiled, not run on hardware
//
#include <hip/hip_runtime.h>

typedef _Float16 h16;
typedef __attribute__((ext_vector_type(16))) _Float16 v16h;
typedef __attribute__((ext_vector_type(8)))  float    v8f;
typedef __attribute__((ext_vector_type(4)))  unsigned int v4u;
typedef __attribute__((ext_vector_type(8)))  int      v8i;
typedef __attribute__((ext_vector_type(4)))  int      v4i;

#if defined(__has_builtin)
#if __has_builtin(__builtin_amdgcn_tensor_load_to_lds) && \
    __has_builtin(__builtin_amdgcn_s_wait_tensorcnt)
#define DSFD_TDM_OK 1
#endif
#endif

// ---------------------------------------------------------------------------
// fp32 -> f16 conversion (input staging)
// ---------------------------------------------------------------------------
__global__ void f32_to_h16_k(const float* __restrict__ s, h16* __restrict__ d, long n) {
    long i  = (long)blockIdx.x * blockDim.x + threadIdx.x;
    long st = (long)gridDim.x * blockDim.x;
    for (; i < n; i += st) d[i] = (h16)s[i];
}

__global__ void zero_f32_k(float* __restrict__ d, long n) {
    long i  = (long)blockIdx.x * blockDim.x + threadIdx.x;
    long st = (long)gridDim.x * blockDim.x;
    for (; i < n; i += st) d[i] = 0.f;
}

// ---------------------------------------------------------------------------
// Weight pre-swizzle: fp32 OIHW -> f16 tiles laid out exactly as the WMMA
// A-fragment wants them in VGPRs (aligned 32-byte vector per lane).
// ---------------------------------------------------------------------------
__global__ void repackA_k(const float* __restrict__ W32, h16* __restrict__ Wp,
                          int Cout, int Kdim, int numM16, int numKb)
{
    const long n = (long)numM16 * numKb * 512;
    long i  = (long)blockIdx.x * blockDim.x + threadIdx.x;
    long st = (long)gridDim.x * blockDim.x;
    for (; i < n; i += st) {
        int  e    = (int)(i & 15);
        int  lane = (int)((i >> 4) & 31);
        long t    = i >> 9;
        int  kb   = (int)(t % numKb);
        int  m16  = (int)(t / numKb);
        int  half = lane >> 4;
        int  lm   = lane & 15;
        int  oc   = m16 * 16 + lm;
        int  k    = kb * 32 + 16 * half + e;
        h16  v    = (h16)0.f;
        if (oc < Cout && k < Kdim) v = (h16)W32[(long)oc * Kdim + k];
        Wp[i] = v;
    }
}

// ---------------------------------------------------------------------------
// Implicit-GEMM convolution via V_WMMA_F32_16X16X32_F16, M-blocked.
//   One wave computes MT*16 (Cout) x 16 (spatial) outputs: per K-step (32),
//   one im2col B gather, MT aligned A vector loads, MT WMMAs (B reused).
//   Waves of a block share the weight stream (same mT) -> WGP$ hits on A.
//   Fast path for 1x1/stride-1/pad-0 convs (k == input channel).
// ---------------------------------------------------------------------------
template <int MT>
__global__ void __launch_bounds__(256)
conv_wmma_k(const h16* __restrict__ X, const h16* __restrict__ Wp,
            const float* __restrict__ Bs,
            h16* __restrict__ Yh, float* __restrict__ Yf,
            int Cin, int H, int W, int Cout, int Ho, int Wo,
            int KH, int KW, int stride, int pad, int dil, int doRelu)
{
    const int lane = threadIdx.x & 31;
    const int wv   = threadIdx.x >> 5;
    const int half = lane >> 4;
    const int lm   = lane & 15;

    const long HWo  = (long)Ho * Wo;
    const int  numM = (Cout + MT * 16 - 1) / (MT * 16);
    const long numN = (HWo + 15) >> 4;
    const long tile = (long)blockIdx.x * 8 + wv;
    if (tile >= (long)numM * numN) return;

    const long nT = tile % numN;
    const int  mT = (int)(tile / numN);

    const int  Kdim  = Cin * KH * KW;
    const int  numKb = (Kdim + 31) >> 5;
    const bool is1x1 = (KH == 1 && KW == 1 && stride == 1 && pad == 0);

    const h16* pA[MT];
#pragma unroll
    for (int s = 0; s < MT; ++s)
        pA[s] = Wp + (((long)(mT * MT + s) * numKb) * 32 + lane) * 16;

    const long n0 = nT * 16 + lm;
    const bool nv = (n0 < HWo);
    const int  oh = nv ? (int)(n0 / Wo) : 0;
    const int  ow = nv ? (int)(n0 % Wo) : 0;
    const int  ihb = oh * stride - pad;
    const int  iwb = ow * stride - pad;
    const long HWin = (long)H * W;

    v8f acc[MT] = {};

#pragma unroll 2
    for (int kb = 0; kb < numKb; ++kb) {
        __builtin_prefetch(pA[0] + (long)(kb + 2) * 512, 0, 1);

        v16h b;
        const int k0 = kb * 32 + 16 * half;
        if (is1x1) {
#pragma unroll
            for (int e = 0; e < 16; ++e) {
                int k = k0 + e;
                b[e] = (nv && k < Kdim) ? X[(long)k * HWin + n0] : (h16)0.f;
            }
        } else {
            int ic = k0 / (KH * KW);
            int r  = k0 % (KH * KW);
            int kh = r / KW;
            int kw = r % KW;
#pragma unroll
            for (int e = 0; e < 16; ++e) {
                h16 v = (h16)0.f;
                int k = k0 + e;
                if (nv && k < Kdim) {
                    int ih = ihb + kh * dil;
                    int iw = iwb + kw * dil;
                    if ((unsigned)ih < (unsigned)H && (unsigned)iw < (unsigned)W)
                        v = X[(long)ic * HWin + (long)ih * W + iw];
                }
                b[e] = v;
                if (++kw == KW) { kw = 0; if (++kh == KH) { kh = 0; ++ic; } }
            }
        }

#pragma unroll
        for (int s = 0; s < MT; ++s) {
            v16h a = *(const v16h*)(pA[s] + (long)kb * 512);
            acc[s] = __builtin_amdgcn_wmma_f32_16x16x32_f16(
                false, a, false, b, (short)0, acc[s], false, false);
        }
    }

    if (nv) {
#pragma unroll
        for (int s = 0; s < MT; ++s) {
#pragma unroll
            for (int r = 0; r < 8; ++r) {
                int oc = mT * (MT * 16) + s * 16 + r + 8 * half;
                if (oc < Cout) {
                    float v = acc[s][r] + Bs[oc];
                    if (doRelu) v = fmaxf(v, 0.f);
                    long off = (long)oc * HWo + n0;
                    if (Yh) Yh[off] = (h16)v;
                    if (Yf) Yf[off] = v;
                }
            }
        }
    }
}

// ---------------------------------------------------------------------------
// 2x2 max pooling (stride 2), NCHW f16
// ---------------------------------------------------------------------------
__global__ void maxpool2_k(const h16* __restrict__ X, h16* __restrict__ Y,
                           int C, int H, int W)
{
    const int Ho = H >> 1, Wo = W >> 1;
    const long n = (long)C * Ho * Wo;
    long i  = (long)blockIdx.x * blockDim.x + threadIdx.x;
    long st = (long)gridDim.x * blockDim.x;
    for (; i < n; i += st) {
        int  wo = (int)(i % Wo);
        long t  = i / Wo;
        int  ho = (int)(t % Ho);
        int  c  = (int)(t / Ho);
        const h16* pr = X + ((long)c * H + 2 * ho) * W + 2 * wo;
        float m = fmaxf(fmaxf((float)pr[0], (float)pr[1]),
                        fmaxf((float)pr[W], (float)pr[W + 1]));
        Y[i] = (h16)m;
    }
}

// ---------------------------------------------------------------------------
// Align-corners bilinear upsample of T (C,th,tw) to (C,H,W), * L, ReLU.
// ---------------------------------------------------------------------------
__global__ void upsmul_relu_k(const h16* __restrict__ T, int th, int tw,
                              const h16* __restrict__ L, h16* __restrict__ Y,
                              int C, int H, int W)
{
    const long n = (long)C * H * W;
    long i  = (long)blockIdx.x * blockDim.x + threadIdx.x;
    long st = (long)gridDim.x * blockDim.x;
    for (; i < n; i += st) {
        int  x = (int)(i % W);
        long t = i / W;
        int  y = (int)(t % H);
        int  c = (int)(t / H);
        float ys = (H > 1) ? (float)y * (float)(th - 1) / (float)(H - 1) : 0.f;
        float xs = (W > 1) ? (float)x * (float)(tw - 1) / (float)(W - 1) : 0.f;
        int y0 = (int)floorf(ys); y0 = y0 < 0 ? 0 : (y0 > th - 2 ? th - 2 : y0);
        int x0 = (int)floorf(xs); x0 = x0 < 0 ? 0 : (x0 > tw - 2 ? tw - 2 : x0);
        float wy = ys - (float)y0;
        float wx = xs - (float)x0;
        const h16* base = T + (long)c * th * tw;
        float g00 = (float)base[(long)y0 * tw + x0];
        float g01 = (float)base[(long)y0 * tw + x0 + 1];
        float g10 = (float)base[(long)(y0 + 1) * tw + x0];
        float g11 = (float)base[(long)(y0 + 1) * tw + x0 + 1];
        float g = g00 * (1.f - wy) * (1.f - wx) + g10 * wy * (1.f - wx)
                + g01 * (1.f - wy) * wx        + g11 * wy * wx;
        float o = g * (float)L[i];
        Y[i] = (h16)fmaxf(o, 0.f);
    }
}

// ---------------------------------------------------------------------------
// Per-pixel channel L2 normalization with learned per-channel gain (in place)
// ---------------------------------------------------------------------------
__global__ void l2norm_k(h16* __restrict__ Xp, const float* __restrict__ g,
                         int C, long HW)
{
    long i  = (long)blockIdx.x * blockDim.x + threadIdx.x;
    long st = (long)gridDim.x * blockDim.x;
    for (; i < HW; i += st) {
        float s = 0.f;
        for (int c = 0; c < C; ++c) {
            float v = (float)Xp[(long)c * HW + i];
            s += v * v;
        }
        float inv = 1.f / (sqrtf(s) + 1e-10f);
        for (int c = 0; c < C; ++c) {
            float v = (float)Xp[(long)c * HW + i];
            Xp[(long)c * HW + i] = (h16)(v * inv * g[c]);
        }
    }
}

// ---------------------------------------------------------------------------
// Decode: priors computed in-kernel, loc -> corner boxes, conf -> softmax[:,1]
// with confidence threshold 0.5.
// ---------------------------------------------------------------------------
#define NPRIORS 21840
__global__ void decode_k(const float* __restrict__ loc,
                         const float* __restrict__ cf,
                         float* __restrict__ boxes,
                         float* __restrict__ scores)
{
    int p = blockIdx.x * blockDim.x + threadIdx.x;
    if (p >= NPRIORS) return;
    const int   starts[7] = {0, 16384, 20480, 21504, 21760, 21824, 21840};
    const int   fs[6]     = {128, 64, 32, 16, 8, 4};
    const float stp[6]    = {4.f, 8.f, 16.f, 32.f, 64.f, 128.f};
    const float ms[6]     = {16.f, 32.f, 64.f, 128.f, 256.f, 512.f};
    int s = 0;
    while (s < 5 && p >= starts[s + 1]) ++s;
    const int  j  = p - starts[s];
    const int  f  = fs[s];
    const long hw = (long)f * f;
    const int  oh = j / f, ow = j % f;

    const float* lb = loc + (long)4 * starts[s];
    float l0 = lb[0 * hw + j], l1 = lb[1 * hw + j];
    float l2 = lb[2 * hw + j], l3 = lb[3 * hw + j];
    const float* cb = cf + (long)2 * starts[s];
    float c0 = cb[j], c1 = cb[hw + j];

    float sc = 1.f / (1.f + expf(c0 - c1));  // softmax[:,1]
    if (!(sc > 0.5f)) sc = 0.f;

    float cx = (ow + 0.5f) * stp[s] / 512.f;
    float cy = (oh + 0.5f) * stp[s] / 512.f;
    float pw = ms[s] / 512.f, ph = ms[s] / 512.f;
    float bcx = cx + l0 * 0.1f * pw;
    float bcy = cy + l1 * 0.1f * ph;
    float bw  = pw * expf(l2 * 0.2f);
    float bh  = ph * expf(l3 * 0.2f);
    boxes[(long)p * 4 + 0] = bcx - bw * 0.5f;
    boxes[(long)p * 4 + 1] = bcy - bh * 0.5f;
    boxes[(long)p * 4 + 2] = bcx + bw * 0.5f;
    boxes[(long)p * 4 + 3] = bcy + bh * 0.5f;
    scores[p] = sc;
}

// ---------------------------------------------------------------------------
// Bitonic sort of 32768 (score,index) pairs, descending, stable tie-break on
// index.  256 KB dynamic LDS (CDNA5: 320 KB / workgroup).  The score tile is
// staged into LDS by the Tensor Data Mover (one TENSOR_LOAD_TO_LDS issued by
// wave 0, waited with s_wait_tensorcnt) when the TDM builtin is available.
// ---------------------------------------------------------------------------
__global__ void __launch_bounds__(1024)
sort_k(const float* __restrict__ sc, float* __restrict__ oS, int* __restrict__ oI)
{
    const int N = 32768;
    extern __shared__ char sm[];
    float* ss = (float*)sm;                       // LDS offset 0
    int*   si = (int*)(sm + (size_t)N * sizeof(float));

#if defined(DSFD_TDM_OK)
    if ((threadIdx.x >> 5) == 0) {
        // ---- Tensor DMA descriptor (D#): 1-D tile of NPRIORS fp32 words ---
        const unsigned long long ga = (unsigned long long)(size_t)sc;
        const unsigned D0 = (unsigned)NPRIORS;
        v4u g0;
        g0[0] = 1u;                                   // count=1 (valid user D#)
        g0[1] = 0u;                                   // lds_addr = 0 (ss[] base)
        g0[2] = (unsigned)(ga & 0xffffffffu);         // global_addr[31:0]
        g0[3] = (unsigned)((ga >> 32) & 0x01ffffffu)  // global_addr[56:32]
              | (2u << 30);                           // type = 2 ("image")
        v8i g1;
        g1[0] = (int)(2u << 16);                      // data_size=2 -> 4 bytes
        g1[1] = (int)((D0 & 0xffffu) << 16);          // tensor_dim0[15:0]
        g1[2] = (int)(((D0 >> 16) & 0xffffu)          // tensor_dim0[31:16]
              | (1u << 16));                          // tensor_dim1 = 1
        g1[3] = (int)((D0 & 0xffffu) << 16);          // tile_dim0 = NPRIORS
        g1[4] = 0;                                    // tile_dim1/2 unused
        g1[5] = (int)D0;                              // tensor_dim0_stride lo
        g1[6] = 0;
        g1[7] = 0;
        v4i z4 = {};
#if __clang_major__ >= 23
        v8i z8 = {};
        __builtin_amdgcn_tensor_load_to_lds(g0, g1, z4, z4, z8, 0);
#else
        __builtin_amdgcn_tensor_load_to_lds(g0, g1, z4, z4, 0);
#endif
        __builtin_amdgcn_s_wait_tensorcnt(0);
    }
    for (int i = threadIdx.x; i < N; i += 1024) {
        if (i >= NPRIORS) ss[i] = -1.f;               // pad region
        si[i] = (i < NPRIORS) ? i : 0;
    }
#else
    for (int i = threadIdx.x; i < N; i += 1024) {
        ss[i] = (i < NPRIORS) ? sc[i] : -1.f;
        si[i] = (i < NPRIORS) ? i : 0;
    }
#endif
    __syncthreads();

    for (int k = 2; k <= N; k <<= 1) {
        for (int j = k >> 1; j > 0; j >>= 1) {
            for (int i = threadIdx.x; i < N; i += 1024) {
                int ixj = i ^ j;
                if (ixj > i) {
                    float a = ss[i], b = ss[ixj];
                    int   ia = si[i], ib = si[ixj];
                    bool before = (a > b) || (a == b && ia < ib);
                    bool wantSwap = ((i & k) == 0) ? !before : before;
                    if (wantSwap) {
                        ss[i] = b; ss[ixj] = a;
                        si[i] = ib; si[ixj] = ia;
                    }
                }
            }
            __syncthreads();
        }
    }
    for (int i = threadIdx.x; i < 5000; i += 1024) {
        oS[i] = ss[i];
        oI[i] = si[i];
    }
}

// ---------------------------------------------------------------------------
// Greedy NMS over the top-5000 sorted boxes, all state in LDS (~105 KB).
// ---------------------------------------------------------------------------
__global__ void __launch_bounds__(1024)
nms_k(const float* __restrict__ boxes, const int* __restrict__ sI,
      unsigned char* __restrict__ keepOut)
{
    const int K = 5000;
    extern __shared__ char sm[];
    float* X1 = (float*)sm;
    float* Y1 = X1 + K;
    float* X2 = Y1 + K;
    float* Y2 = X2 + K;
    float* AR = Y2 + K;
    unsigned char* kp = (unsigned char*)(AR + K);

    for (int i = threadIdx.x; i < K; i += 1024) {
        int id = sI[i];
        float x1 = boxes[(long)id * 4 + 0];
        float y1 = boxes[(long)id * 4 + 1];
        float x2 = boxes[(long)id * 4 + 2];
        float y2 = boxes[(long)id * 4 + 3];
        X1[i] = x1; Y1[i] = y1; X2[i] = x2; Y2[i] = y2;
        AR[i] = (x2 - x1) * (y2 - y1);
        kp[i] = 1;
    }
    __syncthreads();

    for (int i = 0; i < K - 1; ++i) {
        if (kp[i]) {
            float bx1 = X1[i], by1 = Y1[i], bx2 = X2[i], by2 = Y2[i], ba = AR[i];
            for (int j = i + 1 + threadIdx.x; j < K; j += 1024) {
                float iw = fminf(X2[j], bx2) - fmaxf(X1[j], bx1);
                float ih = fminf(Y2[j], by2) - fmaxf(Y1[j], by1);
                iw = fmaxf(iw, 0.f);
                ih = fmaxf(ih, 0.f);
                float inter = iw * ih;
                float iou = inter / (AR[j] + ba - inter);
                if (iou > 0.3f) kp[j] = 0;
            }
        }
        __syncthreads();
    }
    for (int i = threadIdx.x; i < K; i += 1024) keepOut[i] = kp[i];
}

// ---------------------------------------------------------------------------
// Final detection writeout: rank kept boxes, first 750 with score>0.
// Output layout: (1, 2, 750, 5); first half zeros (cleared beforehand).
// ---------------------------------------------------------------------------
__global__ void finalize_k(const float* __restrict__ sS, const int* __restrict__ sI,
                           const unsigned char* __restrict__ kp,
                           const float* __restrict__ boxes, float* __restrict__ out)
{
    if (blockIdx.x != 0 || threadIdx.x != 0) return;
    int r = -1;
    for (int i = 0; i < 5000; ++i) {
        if (kp[i]) {
            ++r;
            float s = sS[i];
            if (r < 750 && s > 0.f) {
                int id = sI[i];
                float* row = out + 3750 + (long)r * 5;
                row[0] = s;
                row[1] = boxes[(long)id * 4 + 0];
                row[2] = boxes[(long)id * 4 + 1];
                row[3] = boxes[(long)id * 4 + 2];
                row[4] = boxes[(long)id * 4 + 3];
            }
        }
    }
}

// ===========================================================================
// Host orchestration
// ===========================================================================
static inline long cdivl(long a, long b) { return (a + b - 1) / b; }
static inline unsigned gblocks(long n, int t) {
    long g = cdivl(n, t);
    if (g < 1) g = 1;
    if (g > 1048576) g = 1048576;
    return (unsigned)g;
}

extern "C" void kernel_launch(void* const* d_in, const int* in_sizes, int n_in,
                              void* d_out, int out_size, void* d_ws, size_t ws_size,
                              hipStream_t stream)
{
    (void)in_sizes; (void)n_in; (void)ws_size;

    // ------------------- input unpacking (setup_inputs dict order) ---------
    int p = 0;
    const float* Xin = (const float*)d_in[p++];
    const float *vw32[15], *vb32[15];
    for (int i = 0; i < 15; ++i) vw32[i] = (const float*)d_in[p++];
    for (int i = 0; i < 15; ++i) vb32[i] = (const float*)d_in[p++];
    const float *ew32[4], *eb32[4];
    for (int i = 0; i < 4; ++i) ew32[i] = (const float*)d_in[p++];
    for (int i = 0; i < 4; ++i) eb32[i] = (const float*)d_in[p++];
    const float *tw32[6], *tb32[6];
    for (int i = 0; i < 6; ++i) tw32[i] = (const float*)d_in[p++];
    for (int i = 0; i < 6; ++i) tb32[i] = (const float*)d_in[p++];
    const float *aw32[5], *ab32[5];
    for (int i = 0; i < 5; ++i) aw32[i] = (const float*)d_in[p++];
    for (int i = 0; i < 5; ++i) ab32[i] = (const float*)d_in[p++];
    const float *fw32[6][6], *fb32[6][6];
    for (int i = 0; i < 6; ++i) for (int j = 0; j < 6; ++j) fw32[i][j] = (const float*)d_in[p++];
    for (int i = 0; i < 6; ++i) for (int j = 0; j < 6; ++j) fb32[i][j] = (const float*)d_in[p++];
    const float* l2g[3];
    for (int i = 0; i < 3; ++i) l2g[i] = (const float*)d_in[p++];
    const float *lw32[6], *lb32[6], *cw32[6], *cb32[6];
    for (int i = 0; i < 6; ++i) lw32[i] = (const float*)d_in[p++];
    for (int i = 0; i < 6; ++i) lb32[i] = (const float*)d_in[p++];
    for (int i = 0; i < 6; ++i) cw32[i] = (const float*)d_in[p++];
    for (int i = 0; i < 6; ++i) cb32[i] = (const float*)d_in[p++];

    // ------------------- workspace bump allocator --------------------------
    size_t off = 0;
    auto alloc = [&](size_t b) -> void* {
        size_t a = (b + 255) & ~(size_t)255;
        char* r = (char*)d_ws + off;
        off += a;
        return r;
    };
    auto AH = [&](long n) { return (h16*)alloc(sizeof(h16) * (size_t)n); };
    auto AF = [&](long n) { return (float*)alloc(sizeof(float) * (size_t)n); };
    auto toh = [&](const float* s, long n) -> h16* {
        h16* d = AH(n);
        f32_to_h16_k<<<dim3(gblocks(n, 256)), dim3(256), 0, stream>>>(s, d, n);
        return d;
    };
    auto mtFor = [](int cout) { return (cout >= 128) ? 8 : 4; };
    auto repack = [&](const float* w32, int cout, int kdim) -> h16* {
        int mt     = mtFor(cout);
        int numM16 = (int)cdivl(cout, mt * 16) * mt;
        int numKb  = (kdim + 31) / 32;
        long n = (long)numM16 * numKb * 512;
        h16* d = AH(n);
        repackA_k<<<dim3(gblocks(n, 256)), dim3(256), 0, stream>>>(
            w32, d, cout, kdim, numM16, numKb);
        return d;
    };

    static const int VS[15][3] = {{64,3,3},{64,64,3},{128,64,3},{128,128,3},
        {256,128,3},{256,256,3},{256,256,3},{512,256,3},{512,512,3},{512,512,3},
        {512,512,3},{512,512,3},{512,512,3},{1024,512,3},{1024,1024,1}};
    static const int ES[4][3] = {{256,1024,1},{512,256,3},{128,512,1},{256,128,3}};
    static const int TS[6][3] = {{256,256,1},{512,256,1},{1024,512,1},
                                 {512,1024,1},{512,512,1},{256,512,1}};
    static const int LS[5][3] = {{512,512,1},{1024,1024,1},{512,512,1},
                                 {512,512,1},{256,256,1}};
    static const int FEMC[6]   = {256, 512, 512, 1024, 512, 256};
    static const int FSZ[6]    = {128, 64, 32, 16, 8, 4};
    static const int STARTS[6] = {0, 16384, 20480, 21504, 21760, 21824};

    h16* vw[15];
    for (int i = 0; i < 15; ++i)
        vw[i] = repack(vw32[i], VS[i][0], VS[i][1] * VS[i][2] * VS[i][2]);
    h16* ew[4];
    for (int i = 0; i < 4; ++i)
        ew[i] = repack(ew32[i], ES[i][0], ES[i][1] * ES[i][2] * ES[i][2]);
    h16* tw[6];
    for (int i = 0; i < 6; ++i)
        tw[i] = repack(tw32[i], TS[i][0], TS[i][1] * TS[i][2] * TS[i][2]);
    h16* aw[5];
    for (int i = 0; i < 5; ++i)
        aw[i] = repack(aw32[i], LS[i][0], LS[i][1] * LS[i][2] * LS[i][2]);
    h16* fw[6][6];
    for (int i = 0; i < 6; ++i) {
        int c = FEMC[i], i2 = c / 3, i1 = c - 2 * i2;
        const int co[6] = {i2, i2, i2, i1, i1, i1};
        const int ci[6] = {c,  c,  i2, c,  i1, i1};
        for (int j = 0; j < 6; ++j) fw[i][j] = repack(fw32[i][j], co[j], ci[j] * 9);
    }
    h16 *lw[6], *cw[6];
    for (int i = 0; i < 6; ++i) lw[i] = repack(lw32[i], 4, FEMC[i] * 9);
    for (int i = 0; i < 6; ++i) cw[i] = repack(cw32[i], 2, FEMC[i] * 9);

    // ------------------- launch helpers ------------------------------------
    auto conv = [&](const h16* x, const h16* w, const float* b,
                    h16* yh, float* yf,
                    int ci, int h, int wd, int co, int k,
                    int st, int pd, int dl, int rl) {
        int ho = (h  + 2 * pd - dl * (k - 1) - 1) / st + 1;
        int wo = (wd + 2 * pd - dl * (k - 1) - 1) / st + 1;
        int mt = mtFor(co);
        long tiles = cdivl(co, mt * 16) * cdivl((long)ho * wo, 16);
        unsigned blocks = (unsigned)cdivl(tiles, 8);
        if (mt == 8)
            conv_wmma_k<8><<<dim3(blocks), dim3(256), 0, stream>>>(
                x, w, b, yh, yf, ci, h, wd, co, ho, wo, k, k, st, pd, dl, rl);
        else
            conv_wmma_k<4><<<dim3(blocks), dim3(256), 0, stream>>>(
                x, w, b, yh, yf, ci, h, wd, co, ho, wo, k, k, st, pd, dl, rl);
    };
    auto pool = [&](const h16* x, h16* y, int c, int h, int w) {
        long n = (long)c * (h / 2) * (w / 2);
        maxpool2_k<<<dim3(gblocks(n, 256)), dim3(256), 0, stream>>>(x, y, c, h, w);
    };
    auto upsmul = [&](const h16* t, int th, int tww, const h16* l, h16* y,
                      int c, int h, int w) {
        long n = (long)c * h * w;
        upsmul_relu_k<<<dim3(gblocks(n, 256)), dim3(256), 0, stream>>>(
            t, th, tww, l, y, c, h, w);
    };

    // ------------------- VGG backbone --------------------------------------
    h16* xh  = toh(Xin, 3L * 512 * 512);
    h16* c11 = AH(64L * 512 * 512);  conv(xh,  vw[0],  vb32[0],  c11, 0, 3,  512, 512, 64,  3, 1, 1, 1, 1);
    h16* c12 = AH(64L * 512 * 512);  conv(c11, vw[1],  vb32[1],  c12, 0, 64, 512, 512, 64,  3, 1, 1, 1, 1);
    h16* p1  = AH(64L * 256 * 256);  pool(c12, p1, 64, 512, 512);
    h16* c21 = AH(128L * 256 * 256); conv(p1,  vw[2],  vb32[2],  c21, 0, 64, 256, 256, 128, 3, 1, 1, 1, 1);
    h16* c22 = AH(128L * 256 * 256); conv(c21, vw[3],  vb32[3],  c22, 0, 128,256, 256, 128, 3, 1, 1, 1, 1);
    h16* p2  = AH(128L * 128 * 128); pool(c22, p2, 128, 256, 256);
    h16* c31 = AH(256L * 128 * 128); conv(p2,  vw[4],  vb32[4],  c31, 0, 128,128, 128, 256, 3, 1, 1, 1, 1);
    h16* c32 = AH(256L * 128 * 128); conv(c31, vw[5],  vb32[5],  c32, 0, 256,128, 128, 256, 3, 1, 1, 1, 1);
    h16* of1 = AH(256L * 128 * 128); conv(c32, vw[6],  vb32[6],  of1, 0, 256,128, 128, 256, 3, 1, 1, 1, 1);
    h16* p3  = AH(256L * 64 * 64);   pool(of1, p3, 256, 128, 128);
    h16* c41 = AH(512L * 64 * 64);   conv(p3,  vw[7],  vb32[7],  c41, 0, 256, 64,  64, 512, 3, 1, 1, 1, 1);
    h16* c42 = AH(512L * 64 * 64);   conv(c41, vw[8],  vb32[8],  c42, 0, 512, 64,  64, 512, 3, 1, 1, 1, 1);
    h16* of2 = AH(512L * 64 * 64);   conv(c42, vw[9],  vb32[9],  of2, 0, 512, 64,  64, 512, 3, 1, 1, 1, 1);
    h16* p4  = AH(512L * 32 * 32);   pool(of2, p4, 512, 64, 64);
    h16* c51 = AH(512L * 32 * 32);   conv(p4,  vw[10], vb32[10], c51, 0, 512, 32,  32, 512, 3, 1, 1, 1, 1);
    h16* c52 = AH(512L * 32 * 32);   conv(c51, vw[11], vb32[11], c52, 0, 512, 32,  32, 512, 3, 1, 1, 1, 1);
    h16* of3 = AH(512L * 32 * 32);   conv(c52, vw[12], vb32[12], of3, 0, 512, 32,  32, 512, 3, 1, 1, 1, 1);
    h16* p5  = AH(512L * 16 * 16);   pool(of3, p5, 512, 32, 32);
    h16* c6  = AH(1024L * 16 * 16);  conv(p5,  vw[13], vb32[13], c6,  0, 512, 16,  16, 1024,3, 1, 3, 3, 1);
    h16* of4 = AH(1024L * 16 * 16);  conv(c6,  vw[14], vb32[14], of4, 0, 1024,16, 16, 1024, 1, 1, 0, 1, 1);

    // ------------------- extras --------------------------------------------
    h16* e0  = AH(256L * 16 * 16);   conv(of4, ew[0], eb32[0], e0,  0, 1024, 16, 16, 256, 1, 1, 0, 1, 1);
    h16* of5 = AH(512L * 8 * 8);     conv(e0,  ew[1], eb32[1], of5, 0, 256,  16, 16, 512, 3, 2, 1, 1, 1);
    h16* e2  = AH(128L * 8 * 8);     conv(of5, ew[2], eb32[2], e2,  0, 512,   8,  8, 128, 1, 1, 0, 1, 1);
    h16* of6 = AH(256L * 4 * 4);     conv(e2,  ew[3], eb32[3], of6, 0, 128,   8,  8, 256, 3, 2, 1, 1, 1);

    // ------------------- top-down path -------------------------------------
    h16* cv7  = AH(256L * 4 * 4);    conv(of6, tw[0], tb32[0], cv7, 0, 256,  4, 4, 256,  1, 1, 0, 1, 1);
    h16* tA   = AH(512L * 4 * 4);    conv(cv7, tw[1], tb32[1], tA,  0, 256,  4, 4, 512,  1, 1, 0, 1, 1);
    h16* lat0 = AH(512L * 8 * 8);    conv(of5, aw[0], ab32[0], lat0,0, 512,  8, 8, 512,  1, 1, 0, 1, 0);
    h16* cv6  = AH(512L * 8 * 8);    upsmul(tA, 4, 4, lat0, cv6, 512, 8, 8);
    h16* tB   = AH(1024L * 8 * 8);   conv(cv6, tw[2], tb32[2], tB,  0, 512,  8, 8, 1024, 1, 1, 0, 1, 1);
    h16* lat1 = AH(1024L * 16 * 16); conv(of4, aw[1], ab32[1], lat1,0, 1024,16,16, 1024, 1, 1, 0, 1, 0);
    h16* fc72 = AH(1024L * 16 * 16); upsmul(tB, 8, 8, lat1, fc72, 1024, 16, 16);
    h16* tC   = AH(512L * 16 * 16);  conv(fc72,tw[3], tb32[3], tC,  0, 1024,16,16, 512,  1, 1, 0, 1, 1);
    h16* lat2 = AH(512L * 32 * 32);  conv(of3, aw[2], ab32[2], lat2,0, 512, 32,32, 512,  1, 1, 0, 1, 0);
    h16* cv5  = AH(512L * 32 * 32);  upsmul(tC, 16, 16, lat2, cv5, 512, 32, 32);
    h16* tD   = AH(512L * 32 * 32);  conv(cv5, tw[4], tb32[4], tD,  0, 512, 32,32, 512,  1, 1, 0, 1, 1);
    h16* lat3 = AH(512L * 64 * 64);  conv(of2, aw[3], ab32[3], lat3,0, 512, 64,64, 512,  1, 1, 0, 1, 0);
    h16* cv4  = AH(512L * 64 * 64);  upsmul(tD, 32, 32, lat3, cv4, 512, 64, 64);
    h16* tE   = AH(256L * 64 * 64);  conv(cv4, tw[5], tb32[5], tE,  0, 512, 64,64, 256,  1, 1, 0, 1, 1);
    h16* lat4 = AH(256L * 128 * 128);conv(of1, aw[4], ab32[4], lat4,0, 256,128,128,256,  1, 1, 0, 1, 0);
    h16* cv3  = AH(256L * 128 * 128);upsmul(tE, 64, 64, lat4, cv3, 256, 128, 128);

    // ------------------- FEM + heads ---------------------------------------
    const h16* srcs[6] = {cv3, cv4, cv5, fc72, cv6, cv7};
    float* locAll = AF(4L * NPRIORS);
    float* cfAll  = AF(2L * NPRIORS);
    for (int i = 0; i < 6; ++i) {
        int  c  = FEMC[i], f = FSZ[i];
        long hw = (long)f * f;
        int  i2 = c / 3, i1 = c - 2 * i2;
        h16* femo = AH((long)c  * hw);
        h16* tA2  = AH((long)i2 * hw);
        h16* tB2  = AH((long)i1 * hw);
        h16* tC2  = AH((long)i1 * hw);
        conv(srcs[i], fw[i][0], fb32[i][0], femo,                 0, c,  f, f, i2, 3, 1, 3, 3, 1);
        conv(srcs[i], fw[i][1], fb32[i][1], tA2,                  0, c,  f, f, i2, 3, 1, 3, 3, 1);
        conv(tA2,     fw[i][2], fb32[i][2], femo + (long)i2 * hw, 0, i2, f, f, i2, 3, 1, 3, 3, 1);
        conv(srcs[i], fw[i][3], fb32[i][3], tB2,                  0, c,  f, f, i1, 3, 1, 3, 3, 1);
        conv(tB2,     fw[i][4], fb32[i][4], tC2,                  0, i1, f, f, i1, 3, 1, 3, 3, 1);
        conv(tC2,     fw[i][5], fb32[i][5], femo + 2L * i2 * hw,  0, i1, f, f, i1, 3, 1, 3, 3, 1);
        if (i < 3)
            l2norm_k<<<dim3(gblocks(hw, 256)), dim3(256), 0, stream>>>(femo, l2g[i], c, hw);
        conv(femo, lw[i], lb32[i], nullptr, locAll + 4L * STARTS[i], c, f, f, 4, 3, 1, 1, 1, 0);
        conv(femo, cw[i], cb32[i], nullptr, cfAll  + 2L * STARTS[i], c, f, f, 2, 3, 1, 1, 1, 0);
    }

    // ------------------- detection -----------------------------------------
    float* boxes  = AF((long)NPRIORS * 4);
    float* scores = AF(NPRIORS);
    decode_k<<<dim3(gblocks(NPRIORS, 256)), dim3(256), 0, stream>>>(
        locAll, cfAll, boxes, scores);

    float* sS = AF(5000);
    int*   sI = (int*)alloc(5000 * sizeof(int));
    sort_k<<<dim3(1), dim3(1024), 32768 * 8, stream>>>(scores, sS, sI);

    unsigned char* kp = (unsigned char*)alloc(5000);
    nms_k<<<dim3(1), dim3(1024), 5 * 5000 * 4 + 5000, stream>>>(boxes, sI, kp);

    zero_f32_k<<<dim3(gblocks(out_size, 256)), dim3(256), 0, stream>>>(
        (float*)d_out, (long)out_size);
    finalize_k<<<dim3(1), dim3(1), 0, stream>>>(sS, sI, kp, boxes, (float*)d_out);
}
